// ESBNLayer_18433999634480
// MI455X (gfx1250) — compile-verified, matrix operand order checked by reading, not run on hardware
//
#include <hip/hip_runtime.h>
#include <hip/hip_bf16.h>
#include <math.h>

typedef __bf16 bf16;
typedef __attribute__((ext_vector_type(16))) bf16  v16bf;
typedef __attribute__((ext_vector_type(8)))  float v8f;

#define T_   512
#define B_   32
#define D_   768
#define H_   512
#define K_   256
#define NWG_ 16
#define HPKE (16 * 2 * 32 * 16)   // elems per hpk ping-pong buffer

// WMMA 16-bit operand layout: lane l = (half<<4)|idx16 ; element e of the 16
// packed bf16 maps to K = e + 8*half + (e>=8 ? 8 : 0)  (ISA 7.12.2 tables).
__device__ __forceinline__ int kmap(int e, int h) { return e + 8*h + ((e >= 8) ? 8 : 0); }

__device__ __forceinline__ v8f wmma_bf16(v16bf a, v16bf b, v8f c) {
  return __builtin_amdgcn_wmma_f32_16x16x32_bf16(false, a, false, b, (short)0, c, false, false);
}

__device__ __forceinline__ float sigmoidf_(float x) { return 1.f / (1.f + __expf(-x)); }

// ---------------- K0: weight pack (bf16, WMMA B-operand layout) + state zero ----
__global__ void init_pack_kernel(const float* __restrict__ Wx,
                                 const float* __restrict__ Wh,
                                 const float* __restrict__ Wk,
                                 bf16* wxpk, bf16* whpk, bf16* wkpk,
                                 bf16* krpk, bf16* hpk, unsigned* cnt) {
  long idx = (long)blockIdx.x * blockDim.x + threadIdx.x;
  const long NWX = 589824;   // 9*128*32*16   (K=257 pad 288, N=2048)
  const long NWH = 1048576;  // 16*128*32*16  (K=512,        N=2048)
  const long NWK = 131072;   // 16*16*32*16   (K=512,        N=256)
  const long NKR = 9216;     // 9*2*32*16
  const long NHP = 2 * HPKE;
  if (idx == 0) { cnt[0] = 0u; cnt[1] = 0u; }
  if (idx < NWX) {
    int e = idx & 15, l = (idx >> 4) & 31; long r = idx >> 9;
    int nt = (int)(r % 128), kt = (int)(r / 128);
    int k = 32 * kt + kmap(e, l >> 4), n = 16 * nt + (l & 15);
    wxpk[idx] = (k < 257) ? (bf16)Wx[k * 2048 + n] : (bf16)0.f;
  } else if (idx < NWX + NWH) {
    long j = idx - NWX;
    int e = j & 15, l = (j >> 4) & 31; long r = j >> 9;
    int nt = (int)(r % 128), kt = (int)(r / 128);
    int k = 32 * kt + kmap(e, l >> 4), n = 16 * nt + (l & 15);
    whpk[j] = (bf16)Wh[k * 2048 + n];
  } else if (idx < NWX + NWH + NWK) {
    long j = idx - NWX - NWH;
    int e = j & 15, l = (j >> 4) & 31; long r = j >> 9;
    int nt = (int)(r % 16), kt = (int)(r / 16);
    int k = 32 * kt + kmap(e, l >> 4), n = 16 * nt + (l & 15);
    wkpk[j] = (bf16)Wk[k * 256 + n];
  } else if (idx < NWX + NWH + NWK + NKR) {
    krpk[idx - (NWX + NWH + NWK)] = (bf16)0.f;        // key_r(step0) = 0
  } else if (idx < NWX + NWH + NWK + NKR + NHP) {
    hpk[idx - (NWX + NWH + NWK + NKR)] = (bf16)0.f;   // h(step0) = 0
  }
}

// ---------------- K1: Gram matrix  S[b] = X[b] @ X[b]^T  via bf16 WMMA ---------
__global__ void __launch_bounds__(256) gram_kernel(const float* __restrict__ X,
                                                   float* __restrict__ SW) {
  int l = threadIdx.x & 31, wv = threadIdx.x >> 5;
  int tile = blockIdx.x * 8 + wv;        // 32 b * 32 itiles * 32 jtiles
  int b  = tile >> 10;
  int it = (tile >> 5) & 31;
  int jt = tile & 31;
  int h = l >> 4, m = l & 15;
  v8f acc = {};
  const float* Arow = X + ((size_t)(b * 512 + it * 16 + m)) * 768;
  const float* Brow = X + ((size_t)(b * 512 + jt * 16 + m)) * 768;
  for (int k0 = 0; k0 < 768; k0 += 32) {
    v16bf a, bb;
    const float* pa = Arow + k0 + 8 * h;  // contiguous runs: K=[8h..8h+7], [16+8h..]
    const float* pb = Brow + k0 + 8 * h;
#pragma unroll
    for (int j = 0; j < 8; ++j) {
      a[j]  = (bf16)pa[j];  a[8 + j]  = (bf16)pa[16 + j];
      bb[j] = (bf16)pb[j];  bb[8 + j] = (bf16)pb[16 + j];
    }
    acc = wmma_bf16(a, bb, acc);
  }
  float* out = SW + ((size_t)b * 512) * 512;
#pragma unroll
  for (int v = 0; v < 8; ++v)
    out[(size_t)(it * 16 + v + 8 * h) * 512 + jt * 16 + m] = acc[v];
}

// ---------------- K2: in-place masked softmax rows + confidence precompute -----
__global__ void __launch_bounds__(256) softmax_kernel(float* __restrict__ SW,
                                                      float* __restrict__ pre_c,
                                                      const float* gain_p,
                                                      const float* bias_p) {
  int bid = blockIdx.x, b = bid >> 9, t = bid & 511, tid = threadIdx.x;
  float* row = SW + ((size_t)(b * 512 + t)) * 512;
  __shared__ float red[256];
  float v0 = (tid < t)       ? row[tid]       : -INFINITY;
  float v1 = (tid + 256 < t) ? row[tid + 256] : -INFINITY;
  red[tid] = fmaxf(v0, v1);
  __syncthreads();
  for (int s = 128; s > 0; s >>= 1) { if (tid < s) red[tid] = fmaxf(red[tid], red[tid + s]); __syncthreads(); }
  float mx = red[0]; __syncthreads();
  float e0 = (tid < t)       ? __expf(v0 - mx) : 0.f;
  float e1 = (tid + 256 < t) ? __expf(v1 - mx) : 0.f;
  red[tid] = e0 + e1;
  __syncthreads();
  for (int s = 128; s > 0; s >>= 1) { if (tid < s) red[tid] += red[tid + s]; __syncthreads(); }
  float inv = (t > 0) ? 1.f / red[0] : 0.f; __syncthreads();
  float gain = *gain_p, bias = *bias_p;
  float w0 = e0 * inv, w1 = e1 * inv, pc = 0.f;
  if (tid < t)       { row[tid]       = w0; pc += w0 * sigmoidf_(v0 * gain + bias); }
  if (tid + 256 < t) { row[tid + 256] = w1; pc += w1 * sigmoidf_(v1 * gain + bias); }
  red[tid] = pc;
  __syncthreads();
  for (int s = 128; s > 0; s >>= 1) { if (tid < s) red[tid] += red[tid + s]; __syncthreads(); }
  if (tid == 0) pre_c[b * 512 + t] = red[0];
}

// ---------------- global barrier helpers (persistent grid, monotonic epoch) ----
__device__ __forceinline__ void gbar(unsigned* cnt, unsigned target) {
  __threadfence();
  __syncthreads();
  if (threadIdx.x == 0) {
    __hip_atomic_fetch_add(cnt, 1u, __ATOMIC_ACQ_REL, __HIP_MEMORY_SCOPE_AGENT);
    while (__hip_atomic_load(cnt, __ATOMIC_ACQUIRE, __HIP_MEMORY_SCOPE_AGENT) < target)
      __builtin_amdgcn_s_sleep(2);
  }
  __syncthreads();
  __threadfence();
}
__device__ __forceinline__ void gwait(unsigned* cnt, unsigned target) {  // observe-only
  __syncthreads();
  if (threadIdx.x == 0) {
    while (__hip_atomic_load(cnt, __ATOMIC_ACQUIRE, __HIP_MEMORY_SCOPE_AGENT) < target)
      __builtin_amdgcn_s_sleep(2);
  }
  __syncthreads();
  __threadfence();
}

// ---------------- K3: persistent 512-step scan --------------------------------
// WGs 0..7 : z-GEMM (256 of 2048 cols each) + gates + h pack + key_w GEMM
// WGs 8..15: attention readout (overlapped with z-GEMM) + g + key_r pack
__global__ void __launch_bounds__(256) scan_kernel(
    const bf16* __restrict__ wxpk, const bf16* __restrict__ whpk,
    const bf16* __restrict__ wkpk,
    bf16* krpk, bf16* hpk,                       // hpk: two ping-pong buffers
    const float* __restrict__ bl, const float* __restrict__ bkey,
    const float* __restrict__ Wg, const float* __restrict__ bg,
    const float* __restrict__ SW, const float* __restrict__ pre_c,
    float* gpart, float* Mk, float* dout, unsigned* cnt1, unsigned* cnt2) {
  const int wg = blockIdx.x, tid = threadIdx.x;
  const int l = tid & 31, wv = tid >> 5;
  const int hh = l >> 4, m15 = l & 15;
  __shared__ float zs[4][32][64];   // z tiles per gate for this WG's 64 h-cols
  __shared__ float cst[32][64];     // persistent LSTM cell state chunk
  __shared__ float pgl[32];         // per-batch partial of h . W_g
  __shared__ float gl[32];          // g per batch (readout WGs)
  if (wg < 8) for (int i = tid; i < 2048; i += 256) ((float*)cst)[i] = 0.f;
  __syncthreads();
  const v16bf* WX = (const v16bf*)wxpk;
  const v16bf* WH = (const v16bf*)whpk;
  const v16bf* WK = (const v16bf*)wkpk;
  unsigned b1t = 0, b2t = 0;
  for (int t = 0; t < 512; ++t) {
    const int rp = t & 1, wp = rp ^ 1;
    b1t += 8; b2t += 16;
    if (wg < 8) {
      // ---- Phase A: z = key_r@Wx + h@Wh ; wave owns 2 N-tiles x 2 M-tiles ----
      const v16bf* KR  = (const v16bf*)krpk;
      const v16bf* HPr = (const v16bf*)hpk + (size_t)rp * (HPKE / 16);
      const int gate = wv >> 1, ps = wv & 1;
      const int nt0 = 32 * gate + 4 * wg + 2 * ps;
      v8f c00 = {}, c01 = {}, c10 = {}, c11 = {};
#pragma unroll 3
      for (int kt = 0; kt < 9; ++kt) {
        v16bf a0 = KR[(kt * 2 + 0) * 32 + l];
        v16bf a1 = KR[(kt * 2 + 1) * 32 + l];
        v16bf bb0 = WX[((size_t)(kt * 128 + nt0)) * 32 + l];
        v16bf bb1 = WX[((size_t)(kt * 128 + nt0 + 1)) * 32 + l];
        c00 = wmma_bf16(a0, bb0, c00); c01 = wmma_bf16(a0, bb1, c01);
        c10 = wmma_bf16(a1, bb0, c10); c11 = wmma_bf16(a1, bb1, c11);
      }
#pragma unroll 4
      for (int kt = 0; kt < 16; ++kt) {
        v16bf a0 = HPr[(kt * 2 + 0) * 32 + l];
        v16bf a1 = HPr[(kt * 2 + 1) * 32 + l];
        v16bf bb0 = WH[((size_t)(kt * 128 + nt0)) * 32 + l];
        v16bf bb1 = WH[((size_t)(kt * 128 + nt0 + 1)) * 32 + l];
        c00 = wmma_bf16(a0, bb0, c00); c01 = wmma_bf16(a0, bb1, c01);
        c10 = wmma_bf16(a1, bb0, c10); c11 = wmma_bf16(a1, bb1, c11);
      }
      if (tid < 32) pgl[tid] = 0.f;
#pragma unroll
      for (int v = 0; v < 8; ++v) {
        int lc = 32 * ps + m15;
        zs[gate][v + 8 * hh][lc]           = c00[v];
        zs[gate][16 + v + 8 * hh][lc]      = c10[v];
        zs[gate][v + 8 * hh][lc + 16]      = c01[v];
        zs[gate][16 + v + 8 * hh][lc + 16] = c11[v];
      }
      __syncthreads();
      // ---- gates / cell / hidden; pack h ; partial g dot ----
      bf16* hpw = hpk + (size_t)wp * HPKE;
#pragma unroll
      for (int q = 0; q < 8; ++q) {
        int idx = tid + q * 256;
        int b = idx >> 6, nc = idx & 63, col = wg * 64 + nc;
        float zi = zs[0][b][nc] + bl[col];
        float zf = zs[1][b][nc] + bl[512 + col];
        float zg = zs[2][b][nc] + bl[1024 + col];
        float zo = zs[3][b][nc] + bl[1536 + col];
        float cv = sigmoidf_(zf) * cst[b][nc] + sigmoidf_(zi) * tanhf(zg);
        cst[b][nc] = cv;
        float hv = sigmoidf_(zo) * tanhf(cv);
        if (t == 511) dout[b * 512 + col] = hv;
        atomicAdd(&pgl[b], hv * Wg[col]);
        int kt = wg * 2 + (nc >> 5), kk = nc & 31;
        int e = (kk & 7) + ((kk >= 16) ? 8 : 0);
        int lane = (((kk >> 3) & 1) << 4) | (b & 15);
        hpw[(((size_t)(kt * 2 + (b >> 4))) * 32 + lane) * 16 + e] = (bf16)hv;
      }
      __syncthreads();
      if (tid < 32) gpart[wg * 32 + tid] = pgl[tid];
      gbar(cnt1, b1t);                 // B1: hpk[wp] + gpart complete (8 WGs)
      // ---- key_w = h @ W_key -> Mk[t] ----
      if (wv < 4) {
        int mt = wv & 1, nt = 2 * wg + (wv >> 1);
        const v16bf* HPw = (const v16bf*)hpk + (size_t)wp * (HPKE / 16);
        v8f acc = {};
#pragma unroll 4
        for (int kt = 0; kt < 16; ++kt) {
          v16bf a  = HPw[(kt * 2 + mt) * 32 + l];
          v16bf bb = WK[(kt * 16 + nt) * 32 + l];
          acc = wmma_bf16(a, bb, acc);
        }
#pragma unroll
        for (int v = 0; v < 8; ++v) {
          int b = 16 * mt + v + 8 * hh, n = 16 * nt + m15;
          Mk[((size_t)t * 32 + b) * 256 + n] = acc[v] + bkey[n];
        }
      }
      gbar(cnt2, b2t);                 // B2: step end (all 16 WGs)
    } else {
      // ---- attention readout, fully overlapped with Phase A of z-WGs ----
      const int rb  = (wg - 8) * 4 + (tid >> 6);
      const int rk0 = (tid & 63) * 4;
      const float* wrow = SW + ((size_t)(rb * 512) + t) * 512;
      const float* mkb  = Mk + rb * 256 + rk0;
      float a0 = 0.f, a1 = 0.f, a2 = 0.f, a3 = 0.f;
#pragma unroll 4
      for (int tp = 0; tp < t; ++tp) {
        __builtin_prefetch((const void*)(mkb + (size_t)(tp + 24) * 8192), 0, 0);
        float w = wrow[tp];
        const float4 mv = *(const float4*)(mkb + (size_t)tp * 8192);
        a0 += w * mv.x; a1 += w * mv.y; a2 += w * mv.z; a3 += w * mv.w;
      }
      gwait(cnt1, b1t);                // observe B1: gpart ready
      if (tid < 32) {
        float s = 0.f;
#pragma unroll
        for (int w8 = 0; w8 < 8; ++w8) s += gpart[w8 * 32 + tid];
        gl[tid] = sigmoidf_(s + bg[0]);
      }
      __syncthreads();
      // ---- key_r_{t+1} = g * [readout, pre_c]  (bf16 A-operand pack) ----
      float gv = gl[rb];
      float vals[4] = { a0 * gv, a1 * gv, a2 * gv, a3 * gv };
#pragma unroll
      for (int j = 0; j < 4; ++j) {
        int k = rk0 + j, kt = k >> 5, kk = k & 31;
        int e = (kk & 7) + ((kk >= 16) ? 8 : 0);
        int lane = (((kk >> 3) & 1) << 4) | (rb & 15);
        krpk[(((size_t)(kt * 2 + (rb >> 4))) * 32 + lane) * 16 + e] = (bf16)vals[j];
      }
      if (wg == 8 && tid < 32) {       // element K (=256): confidence term
        int b = tid;
        float v = gl[b] * pre_c[b * 512 + t];
        krpk[(((size_t)(8 * 2 + (b >> 4))) * 32 + (b & 15)) * 16 + 0] = (bf16)v;
      }
      gbar(cnt2, b2t);                 // B2: step end (all 16 WGs)
    }
  }
}

extern "C" void kernel_launch(void* const* d_in, const int* in_sizes, int n_in,
                              void* d_out, int out_size, void* d_ws, size_t ws_size,
                              hipStream_t stream) {
  const float* X  = (const float*)d_in[0];
  const float* Wx = (const float*)d_in[1];
  const float* Wh = (const float*)d_in[2];
  const float* bl = (const float*)d_in[3];
  const float* Wk = (const float*)d_in[4];
  const float* bk = (const float*)d_in[5];
  const float* Wg = (const float*)d_in[6];
  const float* bg = (const float*)d_in[7];
  const float* cg = (const float*)d_in[8];
  const float* cb = (const float*)d_in[9];
  (void)in_sizes; (void)n_in; (void)out_size; (void)ws_size;
  char* p = (char*)d_ws;
  size_t off = 0;
  auto alloc = [&](size_t bytes) { void* r = p + off; off = (off + bytes + 255) & ~(size_t)255; return r; };
  unsigned* cnt  = (unsigned*)alloc(8);             // cnt1, cnt2
  float*    gprt = (float*)alloc(8 * 32 * 4);
  float*    prec = (float*)alloc(32 * 512 * 4);
  bf16*     krpk = (bf16*)alloc(9 * 2 * 32 * 16 * 2);
  bf16*     hpk  = (bf16*)alloc((size_t)2 * HPKE * 2);
  bf16*     wxpk = (bf16*)alloc((size_t)589824 * 2);
  bf16*     whpk = (bf16*)alloc((size_t)1048576 * 2);
  bf16*     wkpk = (bf16*)alloc((size_t)131072 * 2);
  float*    Mk   = (float*)alloc((size_t)512 * 32 * 256 * 4);
  float*    SW   = (float*)alloc((size_t)32 * 512 * 512 * 4);
  long total0 = 589824 + 1048576 + 131072 + 9216 + 2 * HPKE;
  init_pack_kernel<<<(int)((total0 + 255) / 256), 256, 0, stream>>>(
      Wx, Wh, Wk, wxpk, whpk, wkpk, krpk, hpk, cnt);
  gram_kernel<<<4096, 256, 0, stream>>>(X, SW);
  softmax_kernel<<<32 * 512, 256, 0, stream>>>(SW, prec, cg, cb);
  scan_kernel<<<NWG_, 256, 0, stream>>>(wxpk, whpk, wkpk, krpk, hpk,
                                        bl, bk, Wg, bg, SW, prec,
                                        gprt, Mk, (float*)d_out, cnt, cnt + 1);
}
